// ARNet_43336220017103
// MI455X (gfx1250) — matrix operand order repaired and checked by reading, not compiled, not run on hardware
//
#include <hip/hip_runtime.h>

#define B_  16
#define N_  256
#define D_  12
#define M_  64
#define EI_ 25
#define EH_ 50
#define LIP_ 0.909f

typedef _Float16 v16h __attribute__((ext_vector_type(16)));
typedef _Float16 h8   __attribute__((ext_vector_type(8)));
typedef float    v8f  __attribute__((ext_vector_type(8)));

// sigmoid: prefer gfx1250 v_tanh_f32 (1 trans op); else exp + fast rcp (no IEEE div chain)
__device__ __forceinline__ float sigm(float x) {
#if __has_builtin(__builtin_amdgcn_tanhf)
    return fmaf(0.5f, __builtin_amdgcn_tanhf(x * 0.5f), 0.5f);
#else
    return __builtin_amdgcn_rcpf(1.0f + __expf(-x));
#endif
}
__device__ __forceinline__ float lips(float x) { return LIP_ * x * sigm(x); }
__device__ __forceinline__ v16h hcat(h8 a, h8 b) {
    return __builtin_shufflevector(a, b, 0,1,2,3,4,5,6,7,8,9,10,11,12,13,14,15);
}

// feats0 = concat([x, x], axis=-1)
__global__ void egnn_init_433362(const float* __restrict__ x, float* __restrict__ f0) {
    int idx = blockIdx.x * blockDim.x + threadIdx.x;
    if (idx < B_ * N_ * D_) {
        int d  = idx % D_;
        int bn = idx / D_;
        f0[idx] = x[bn * 6 + (d % 6)];
    }
}

// One block per (b,i) node. 4 waves; each wave owns j-tiles of 16 pairs.
// GEMMs are operand-swapped:  D = W^T x tile^T  so each lane holds one j-row's
// channels in registers -> gate + j-reduction stay in VGPRs.
__global__ __launch_bounds__(128) void egnn_layer_433362(
    const float* __restrict__ x, const unsigned char* __restrict__ mask,
    const float* __restrict__ fin, float* __restrict__ fout,
    const float* __restrict__ ew1, const float* __restrict__ eb1,
    const float* __restrict__ ew2, const float* __restrict__ eb2,
    const float* __restrict__ gw,  const float* __restrict__ gb,
    const float* __restrict__ lng, const float* __restrict__ lnb,
    const float* __restrict__ nw1, const float* __restrict__ nb1,
    const float* __restrict__ nw2, const float* __restrict__ nb2,
    const float* __restrict__ mw1, const float* __restrict__ mb1,
    const float* __restrict__ mw2, const float* __restrict__ mb2,
    float* __restrict__ out, int layer, int is_last)
{
    // weights staged [n_out][k] fp16; k==EI_/EH_ row carries the bias (matching
    // constant-1 column appended to e / h)
    __shared__ __align__(64) _Float16 sW1[64][32];   // edge_w1^T: K=25(+1 bias)->32, N=50->64
    __shared__ __align__(64) _Float16 sW2[64][64];   // edge_w2^T: K=50(+1 bias)->64, N=64
    __shared__ __align__(16) float sGW[64];
    __shared__ __align__(64) _Float16 sE[4][16][32]; // per-wave edge-input tile (rows x K)
    __shared__ __align__(64) _Float16 sH[4][16][64]; // per-wave hidden tile (rows x K)
    __shared__ float sRed[4][32][32];                // per-lane channel accumulators
    __shared__ float sFI[12], sNI[80], sHID[24], sY[64], sO[2];

    const int tid  = threadIdx.x;
    const int wave = tid >> 5;
    const int lane = tid & 31;
    const int bi   = blockIdx.x;
    const int b    = bi >> 8;
    const int i    = bi & (N_ - 1);

    const float* ew1L = ew1 + layer * EI_ * EH_;
    const float* eb1L = eb1 + layer * EH_;
    const float* ew2L = ew2 + layer * EH_ * M_;
    const float* eb2L = eb2 + layer * M_;
    const float* gwL  = gw  + layer * M_;
    const float* lngL = lng + layer * D_;
    const float* lnbL = lnb + layer * D_;
    const float* nw1L = nw1 + layer * (D_ + M_) * (2 * D_);
    const float* nb1L = nb1 + layer * (2 * D_);
    const float* nw2L = nw2 + layer * (2 * D_) * D_;
    const float* nb2L = nb2 + layer * D_;

    // ---- stage weights (transposed, bias row folded, zero-padded) ----
    for (int idx = tid; idx < 64 * 32; idx += 128) {
        int n = idx >> 5, k = idx & 31;
        float v = 0.0f;
        if (n < EH_) {
            if (k < EI_)       v = ew1L[k * EH_ + n];
            else if (k == EI_) v = eb1L[n];
        }
        sW1[n][k] = (_Float16)v;
    }
    for (int idx = tid; idx < 64 * 64; idx += 128) {
        int n = idx >> 6, k = idx & 63;
        float v = (k < EH_) ? ew2L[k * M_ + n] : ((k == EH_) ? eb2L[n] : 0.0f);
        sW2[n][k] = (_Float16)v;
    }
    if (tid < 64) sGW[tid] = gwL[tid];

    float fi[12], xi[6];
    const float* fip = fin + (b * N_ + i) * D_;
    const float* xip = x   + (b * N_ + i) * 6;
    #pragma unroll
    for (int d = 0; d < 12; ++d) fi[d] = fip[d];
    #pragma unroll
    for (int d = 0; d < 6; ++d)  xi[d] = xip[d];
    _Float16 fih[12];
    #pragma unroll
    for (int d = 0; d < 12; ++d) fih[d] = (_Float16)fi[d];
    const float mi_mask = mask[b * N_ + i] ? 1.0f : 0.0f;
    const float gbv = gb[layer];
    __syncthreads();

    const int row  = lane & 15;            // j-row owned by this lane
    const int kb   = (lane < 16) ? 0 : 8;  // A-fragment K-half base
    const int khb  = (lane < 16) ? 0 : 8;  // B-fragment K base
    const int rof8 = (lane < 16) ? 0 : 8;  // channel half offset

    // ---- hoist stationary W1 A-fragments + gate weights into registers ----
    v16h aW1[4];
    #pragma unroll
    for (int t = 0; t < 4; ++t) {
        int n = t * 16 + row;
        h8 lo = *(const h8*)&sW1[n][kb];
        h8 hi = *(const h8*)&sW1[n][kb + 16];
        aW1[t] = hcat(lo, hi);
    }
    float gwr[32];
    #pragma unroll
    for (int t = 0; t < 4; ++t) {
        float4 q0 = *(const float4*)&sGW[t * 16 + rof8];
        float4 q1 = *(const float4*)&sGW[t * 16 + rof8 + 4];
        gwr[t*8+0]=q0.x; gwr[t*8+1]=q0.y; gwr[t*8+2]=q0.z; gwr[t*8+3]=q0.w;
        gwr[t*8+4]=q1.x; gwr[t*8+5]=q1.y; gwr[t*8+6]=q1.z; gwr[t*8+7]=q1.w;
    }

    float macc[32];
    #pragma unroll
    for (int c = 0; c < 32; ++c) macc[c] = 0.0f;

    for (int rt = wave; rt < 16; rt += 4) {
        const int j0 = rt * 16;
        // ---- build e-row = [fi, fj, |xi-xj|^2, 1(bias), pad] (lanes 0-15) ----
        if (lane < 16) {
            const int j = j0 + lane;
            const float* fjp = fin + (b * N_ + j) * D_;
            const float* xjp = x   + (b * N_ + j) * 6;
            float dist = 0.0f;
            #pragma unroll
            for (int d = 0; d < 6; ++d) { float dd = xi[d] - xjp[d]; dist += dd * dd; }
            __align__(16) _Float16 eh[32];
            #pragma unroll
            for (int d = 0; d < 12; ++d) { eh[d] = fih[d]; eh[12 + d] = (_Float16)fjp[d]; }
            eh[24] = (_Float16)dist;
            eh[25] = (_Float16)1.0f;           // bias column
            #pragma unroll
            for (int c = 26; c < 32; ++c) eh[c] = (_Float16)0.0f;
            #pragma unroll
            for (int q = 0; q < 4; ++q)
                *(uint4*)&sE[wave][lane][q * 8] = *(const uint4*)&eh[q * 8];
            if (rt + 4 < 16)
                __builtin_prefetch(fin + (b * N_ + j0 + 64 + lane) * D_, 0, 3);
        }
        // ---- GEMM1 (swapped): D = W1^T[16x32] x e^T[32x16] ----
        h8 e0 = *(const h8*)&sE[wave][row][khb];
        h8 e1 = *(const h8*)&sE[wave][row][khb + 16];
        v16h be = hcat(e0, e1);
        v8f acc1[4];
        #pragma unroll
        for (int t = 0; t < 4; ++t) {
            v8f cz = {};
            acc1[t] = __builtin_amdgcn_wmma_f32_16x16x32_f16(
                false, aW1[t], false, be, (short)0, cz, false, false);
        }
        // ---- lipswish -> packed fp16 hidden row (each lane: 8 contiguous chans) ----
        #pragma unroll
        for (int t = 0; t < 4; ++t) {
            __align__(16) _Float16 hp[8];
            #pragma unroll
            for (int v = 0; v < 8; ++v) {
                float hv = lips(acc1[t][v]);
                if (t == 3 && v == 2) hv = (lane < 16) ? 1.0f : hv;  // h[:,50]=1 bias col
                hp[v] = (_Float16)hv;
            }
            *(uint4*)&sH[wave][row][t * 16 + rof8] = *(const uint4*)&hp[0];
        }
        // ---- GEMM2 (swapped): D = W2^T[16x32] x h^T[32x16], 2 K-steps ----
        v8f acc2[4];
        #pragma unroll
        for (int t = 0; t < 4; ++t) { v8f z = {}; acc2[t] = z; }
        #pragma unroll
        for (int ks = 0; ks < 2; ++ks) {
            h8 hb0 = *(const h8*)&sH[wave][row][ks * 32 + khb];
            h8 hb1 = *(const h8*)&sH[wave][row][ks * 32 + khb + 16];
            v16h bh = hcat(hb0, hb1);
            #pragma unroll
            for (int t = 0; t < 4; ++t) {
                int n = t * 16 + row;
                h8 wlo = *(const h8*)&sW2[n][ks * 32 + kb];
                h8 whi = *(const h8*)&sW2[n][ks * 32 + kb + 16];
                v16h aw2 = hcat(wlo, whi);
                acc2[t] = __builtin_amdgcn_wmma_f32_16x16x32_f16(
                    false, aw2, false, bh, (short)0, acc2[t], false, false);
            }
        }
        // ---- lipswish + gate + gated j-accumulation, all in registers ----
        float gp = 0.0f;
        #pragma unroll
        for (int t = 0; t < 4; ++t) {
            #pragma unroll
            for (int v = 0; v < 8; ++v) {
                float mv = lips(acc2[t][v]);
                acc2[t][v] = mv;
                gp = fmaf(mv, gwr[t * 8 + v], gp);
            }
        }
        float g = gp + __shfl_xor(gp, 16);
        g = sigm(g + gbv) * mi_mask * (mask[b * N_ + j0 + row] ? 1.0f : 0.0f);
        #pragma unroll
        for (int t = 0; t < 4; ++t)
            #pragma unroll
            for (int v = 0; v < 8; ++v)
                macc[t * 8 + v] = fmaf(acc2[t][v], g, macc[t * 8 + v]);
    }

    // ---- dump per-lane accumulators; reduce across lanes+waves ----
    #pragma unroll
    for (int c = 0; c < 32; ++c) sRed[wave][lane][c] = macc[c];
    if (tid < 12) sFI[tid] = fi[tid];
    __syncthreads();
    if (tid < 64) {
        int c = tid, t = c >> 4, half = (c >> 3) & 1, v = c & 7, idx = t * 8 + v;
        float s = 0.0f;
        #pragma unroll
        for (int w = 0; w < 4; ++w)
            for (int L = 0; L < 16; ++L) s += sRed[w][L + half * 16][idx];
        sNI[12 + c] = s;                              // m_i
    }
    if (tid < 12) {                                   // LayerNorm(feats_i)
        float mu = 0.0f, var = 0.0f;
        #pragma unroll
        for (int d = 0; d < 12; ++d) mu += sFI[d];
        mu *= (1.0f / 12.0f);
        #pragma unroll
        for (int d = 0; d < 12; ++d) { float dd = sFI[d] - mu; var += dd * dd; }
        var *= (1.0f / 12.0f);
        sNI[tid] = (sFI[tid] - mu) * rsqrtf(var + 1e-5f) * lngL[tid] + lnbL[tid];
    }
    __syncthreads();
    // ---- node MLP: [76]->[24]->[12] + residual ----
    if (tid < 24) {
        float hs = nb1L[tid];
        for (int k = 0; k < 76; ++k) hs += sNI[k] * nw1L[k * 24 + tid];
        sHID[tid] = lips(hs);
    }
    __syncthreads();
    if (tid < 12) {
        float fn = nb2L[tid] + fi[tid];
        #pragma unroll
        for (int k = 0; k < 24; ++k) fn += sHID[k] * nw2L[k * 12 + tid];
        fout[(b * N_ + i) * D_ + tid] = fn;
        sFI[tid] = fn;
    }
    // ---- final head (layer 1 only): mask, [12]->relu[64]->[2], pad to [2,6] ----
    if (is_last) {
        __syncthreads();
        if (tid < 64) {
            float y = mb1[tid];
            #pragma unroll
            for (int k = 0; k < 12; ++k) y += (sFI[k] * mi_mask) * mw1[k * 64 + tid];
            sY[tid] = fmaxf(y, 0.0f);
        }
        __syncthreads();
        if (tid < 2) {
            float o = mb2[tid];
            #pragma unroll
            for (int k = 0; k < 64; ++k) o += sY[k] * mw2[k * 2 + tid];
            sO[tid] = o;
        }
        __syncthreads();
        if (tid < 12) {
            int ch = tid / 6, p = tid - ch * 6;
            out[(b * N_ + i) * 12 + tid] = (p == 0) ? sO[ch] : 0.0f;
        }
    }
}

extern "C" void kernel_launch(void* const* d_in, const int* in_sizes, int n_in,
                              void* d_out, int out_size, void* d_ws, size_t ws_size,
                              hipStream_t stream)
{
    const float* x            = (const float*)d_in[0];
    const unsigned char* mask = (const unsigned char*)d_in[1];
    const float* ew1 = (const float*)d_in[2];
    const float* eb1 = (const float*)d_in[3];
    const float* ew2 = (const float*)d_in[4];
    const float* eb2 = (const float*)d_in[5];
    const float* gw  = (const float*)d_in[6];
    const float* gb  = (const float*)d_in[7];
    const float* lng = (const float*)d_in[8];
    const float* lnb = (const float*)d_in[9];
    const float* nw1 = (const float*)d_in[10];
    const float* nb1 = (const float*)d_in[11];
    const float* nw2 = (const float*)d_in[12];
    const float* nb2 = (const float*)d_in[13];
    const float* mw1 = (const float*)d_in[14];
    const float* mb1 = (const float*)d_in[15];
    const float* mw2 = (const float*)d_in[16];
    const float* mb2 = (const float*)d_in[17];
    float* out = (float*)d_out;

    float* fa = (float*)d_ws;             // feats buffer A: [B,N,12]
    float* fb = fa + B_ * N_ * D_;        // feats buffer B

    egnn_init_433362<<<(B_ * N_ * D_ + 127) / 128, 128, 0, stream>>>(x, fa);
    egnn_layer_433362<<<B_ * N_, 128, 0, stream>>>(
        x, mask, fa, fb, ew1, eb1, ew2, eb2, gw, gb, lng, lnb,
        nw1, nb1, nw2, nb2, mw1, mb1, mw2, mb2, out, 0, 0);
    egnn_layer_433362<<<B_ * N_, 128, 0, stream>>>(
        x, mask, fb, fa, ew1, eb1, ew2, eb2, gw, gb, lng, lnb,
        nw1, nb1, nw2, nb2, mw1, mb1, mw2, mb2, out, 1, 1);
}
